// ViTBlock_62663572848744
// MI455X (gfx1250) — matrix operand order repaired and checked by reading, not compiled, not run on hardware
//
#include <hip/hip_runtime.h>
#include <math.h>

// ---------------- types ----------------
typedef __attribute__((ext_vector_type(16))) __bf16 v16bf;
typedef __attribute__((ext_vector_type(8)))  __bf16 v8bf;
typedef __attribute__((ext_vector_type(8)))  float  v8f;

__device__ __forceinline__ v8f wmma_bf16(v16bf a, v16bf b, v8f c) {
  // D = A(16x32) * B(32x16) + C, fp32 accum
  return __builtin_amdgcn_wmma_f32_16x16x32_bf16(false, a, false, b, (short)0, c, false, false);
}

// load a 16-element bf16 fragment from two 16B-aligned 8-element chunks
__device__ __forceinline__ v16bf ld16(const __bf16* p0, const __bf16* p1) {
  v8bf a = *(const v8bf*)p0;
  v8bf b = *(const v8bf*)p1;
  v16bf r;
#pragma unroll
  for (int i = 0; i < 8; ++i) { r[i] = a[i]; r[i + 8] = b[i]; }
  return r;
}

// ---------------- async global->LDS staging (CDNA5 GLOBAL_LOAD_ASYNC_TO_LDS) ----------------
#if defined(__has_builtin)
#if __has_builtin(__builtin_amdgcn_global_load_async_to_lds_b128) && \
    __has_builtin(__builtin_amdgcn_s_wait_asynccnt)
#define HAVE_ASYNC_LDS 1
#endif
#endif

#ifdef HAVE_ASYNC_LDS
typedef int v4i_vs __attribute__((vector_size(16)));
typedef __attribute__((address_space(1))) v4i_vs* g_v4i_ptr;
typedef __attribute__((address_space(3))) v4i_vs* l_v4i_ptr;
__device__ __forceinline__ void cp16_g2l(void* lds, const void* glob) {
  __builtin_amdgcn_global_load_async_to_lds_b128(
      (g_v4i_ptr)(void*)glob, (l_v4i_ptr)lds, 0, 0);
}
__device__ __forceinline__ void cp_wait() { __builtin_amdgcn_s_wait_asynccnt(0); }
#else
__device__ __forceinline__ void cp16_g2l(void* lds, const void* glob) {
  *(int4*)lds = *(const int4*)glob;
}
__device__ __forceinline__ void cp_wait() {}
#endif

// ---------------- problem constants ----------------
// B=1, H=W=D=32, C=512, NH=8, HD=64, P=8, N=512 (tokens/window), 64 windows
#define CCH   512
#define NHEAD 8
#define HDIM  64
#define NTOK  32768          // total tokens
#define NWIN  64
#define HID   2048

// window-order row g -> image-order token t
__device__ __forceinline__ int g_to_t(int g) {
  int wid = g >> 9, n = g & 511;
  int h = ((wid >> 4) << 3) + (n >> 6);
  int w = (((wid >> 2) & 3) << 3) + ((n >> 3) & 7);
  int d = ((wid & 3) << 3) + (n & 7);
  return (h << 10) + (w << 5) + d;
}

// ---------------- fp32 -> bf16 convert ----------------
__global__ void cvt_bf16_kernel(const float* __restrict__ in, __bf16* __restrict__ out, int n) {
  int i = blockIdx.x * 256 + threadIdx.x;
  if (i < n) out[i] = (__bf16)in[i];
}

// ---------------- LayerNorm (one wave per token, wave32) ----------------
template <bool WINDOW_ORDER>
__global__ __launch_bounds__(256) void ln_kernel(const float* __restrict__ x,
                                                 const float* __restrict__ w,
                                                 const float* __restrict__ b,
                                                 __bf16* __restrict__ out) {
  const int g    = blockIdx.x * 8 + (threadIdx.x >> 5);
  const int lane = threadIdx.x & 31;
  const int t    = WINDOW_ORDER ? g_to_t(g) : g;
  const float* xp = x + (size_t)t * CCH;

  float4 v[4];
  float s = 0.f;
#pragma unroll
  for (int i = 0; i < 4; ++i) {
    v[i] = *(const float4*)(xp + i * 128 + lane * 4);
    s += v[i].x + v[i].y + v[i].z + v[i].w;
  }
#pragma unroll
  for (int m = 16; m >= 1; m >>= 1) s += __shfl_xor(s, m, 32);
  const float mean = s * (1.0f / CCH);

  float vs = 0.f;
#pragma unroll
  for (int i = 0; i < 4; ++i) {
    float dx;
    dx = v[i].x - mean; vs += dx * dx;
    dx = v[i].y - mean; vs += dx * dx;
    dx = v[i].z - mean; vs += dx * dx;
    dx = v[i].w - mean; vs += dx * dx;
  }
#pragma unroll
  for (int m = 16; m >= 1; m >>= 1) vs += __shfl_xor(vs, m, 32);
  const float rstd = rsqrtf(vs * (1.0f / CCH) + 1e-5f);

  __bf16* op = out + (size_t)g * CCH;
#pragma unroll
  for (int i = 0; i < 4; ++i) {
    const int c = i * 128 + lane * 4;
    const float* vf = (const float*)&v[i];
#pragma unroll
    for (int j = 0; j < 4; ++j)
      op[c + j] = (__bf16)((vf[j] - mean) * rstd * w[c + j] + b[c + j]);
  }
}

// ---------------- generic bf16 WMMA GEMM, 128x128 tile, K-slab 64 ----------------
// EPI: 0=kv(bias->bf16), 1=proj(bias + window->image + shortcut -> f32),
//      2=fc1(bias+gelu->bf16), 3=fc2(bias + residual -> f32)
template <int EPI>
__global__ __launch_bounds__(256) void gemm_kernel(const __bf16* __restrict__ A,
                                                   const __bf16* __restrict__ B,
                                                   const float* __restrict__ bias,
                                                   const float* __restrict__ resid,
                                                   float* __restrict__ outF,
                                                   __bf16* __restrict__ outB,
                                                   int M, int N, int K) {
  __shared__ __attribute__((aligned(16))) __bf16 As [128][72];  // [row][k]
  __shared__ __attribute__((aligned(16))) __bf16 Bst[128][72];  // [n][k] (transposed)

  const int tid = threadIdx.x, wv = tid >> 5, lane = tid & 31;
  const int lh = lane & 15, sel = lane >> 4;
  const int wm = wv & 3, wn = wv >> 2;                 // 4x2 wave grid -> 32x64 per wave
  const int bM = blockIdx.y * 128, bN = blockIdx.x * 128;

  v8f acc[2][4] = {};

  for (int k0 = 0; k0 < K; k0 += 64) {
    __syncthreads();
    // A tile: 128 rows x 64 k — async global->LDS (b128 per lane)
#pragma unroll
    for (int i = 0; i < 4; ++i) {
      int slot = tid + i * 256;               // 1024 slots of 8 bf16
      int row = slot >> 3, kk = (slot & 7) * 8;
      cp16_g2l(&As[row][kk], &A[(size_t)(bM + row) * K + k0 + kk]);
    }
    // B tile: 64 k-rows x 128 n, transposed into Bst[n][k] (VALU path: transform)
#pragma unroll
    for (int i = 0; i < 2; ++i) {
      int slot = tid + i * 256;               // 512 slots of 16 n
      int kr = slot >> 3, nn = (slot & 7) * 16;
      const __bf16* bp = &B[(size_t)(k0 + kr) * N + bN + nn];
      int4 t0 = *(const int4*)bp, t1 = *(const int4*)(bp + 8);
      const __bf16* e0 = (const __bf16*)&t0;
      const __bf16* e1 = (const __bf16*)&t1;
#pragma unroll
      for (int j = 0; j < 8; ++j) { Bst[nn + j][kr] = e0[j]; Bst[nn + 8 + j][kr] = e1[j]; }
    }
    cp_wait();              // ASYNCcnt==0: async LDS writes landed
    __syncthreads();

#pragma unroll
    for (int ks = 0; ks < 2; ++ks) {
      v16bf af[2];
#pragma unroll
      for (int mt = 0; mt < 2; ++mt) {
        const __bf16* ar = &As[wm * 32 + mt * 16 + lh][ks * 32];
        af[mt] = ld16(ar + sel * 8, ar + 16 + sel * 8);
      }
#pragma unroll
      for (int nt = 0; nt < 4; ++nt) {
        const __bf16* br = &Bst[wn * 64 + nt * 16 + lh][ks * 32];
        v16bf bf_ = ld16(br + sel * 8, br + 16 + sel * 8);
        acc[0][nt] = wmma_bf16(af[0], bf_, acc[0][nt]);
        acc[1][nt] = wmma_bf16(af[1], bf_, acc[1][nt]);
      }
    }
  }

  // epilogue (C layout: vgpr r = row slot, lane half = row group, lane%16 = col)
#pragma unroll
  for (int mt = 0; mt < 2; ++mt) {
#pragma unroll
    for (int r = 0; r < 8; ++r) {
      const int row = bM + wm * 32 + mt * 16 + sel * 8 + r;
      int trow = row;
      if (EPI == 1) trow = g_to_t(row);       // proj: window order -> image order
#pragma unroll
      for (int nt = 0; nt < 4; ++nt) {
        const int col = bN + wn * 64 + nt * 16 + lh;
        float v = acc[mt][nt][r] + bias[col];
        if (EPI == 0) {
          outB[(size_t)row * N + col] = (__bf16)v;
        } else if (EPI == 1) {
          outF[(size_t)trow * N + col] = v + resid[(size_t)trow * N + col];
        } else if (EPI == 2) {
          v = 0.5f * v * (1.0f + erff(v * 0.70710678118f));   // exact GELU
          outB[(size_t)row * N + col] = (__bf16)v;
        } else {
          outF[(size_t)row * N + col] = v + resid[(size_t)row * N + col];
        }
      }
    }
  }
}

// ---------------- flash attention: per (window, head, 128-q block) ----------------
__global__ __launch_bounds__(256) void attn_kernel(const float* __restrict__ q_ms,
                                                   const __bf16* __restrict__ kv,
                                                   __bf16* __restrict__ attn_out) {
  __shared__ __attribute__((aligned(16))) __bf16 Qs[128][72];     // [qrow][hd]
  __shared__ __attribute__((aligned(16))) __bf16 Ks[64][72];      // [kvrow][hd]
  __shared__ __attribute__((aligned(16))) __bf16 Vt[64][72];      // [hd][kvrow]
  __shared__ __attribute__((aligned(16))) __bf16 Ps[8][16][72];   // per-wave P tile

  const int qblk = blockIdx.x, head = blockIdx.y, wid = blockIdx.z;
  const int tid = threadIdx.x, wv = tid >> 5, lane = tid & 31;
  const int lh = lane & 15, sel = lane >> 4;

  // stage Q (scaled by HD^-0.5 = 0.125) as bf16
  {
    const float* qp = q_ms + ((size_t)head * 512 + qblk * 128) * HDIM;
#pragma unroll
    for (int i = 0; i < 8; ++i) {
      int slot = tid + i * 256;                 // 2048 float4 slots
      int row = slot >> 4, c4 = (slot & 15) * 4;
      float4 f = *(const float4*)(qp + row * HDIM + c4);
      Qs[row][c4 + 0] = (__bf16)(f.x * 0.125f);
      Qs[row][c4 + 1] = (__bf16)(f.y * 0.125f);
      Qs[row][c4 + 2] = (__bf16)(f.z * 0.125f);
      Qs[row][c4 + 3] = (__bf16)(f.w * 0.125f);
    }
  }
  __syncthreads();

  v16bf qf[2];
  {
    const __bf16* qr = &Qs[wv * 16 + lh][0];
    qf[0] = ld16(qr + sel * 8,      qr + 16 + sel * 8);
    qf[1] = ld16(qr + 32 + sel * 8, qr + 48 + sel * 8);
  }

  float mrun[8], lrun[8];
  v8f O[4] = {};
#pragma unroll
  for (int r = 0; r < 8; ++r) { mrun[r] = -1e30f; lrun[r] = 0.f; }

  const size_t kvbase = ((size_t)wid * 512) * 1024 + (size_t)head * HDIM;

  for (int c0 = 0; c0 < 512; c0 += 64) {       // stream KV in 64-row chunks
    __syncthreads();
#pragma unroll
    for (int i = 0; i < 2; ++i) {
      int slot = tid + i * 256;                // 512 slots of 8 bf16
      int r = slot >> 3, h8 = (slot & 7) * 8;
      // K: natural layout, async global->LDS
      cp16_g2l(&Ks[r][h8], &kv[kvbase + (size_t)(c0 + r) * 1024 + h8]);
      // V: transpose into Vt[hd][kvrow] (VALU path: transform)
      int4 t4 = *(const int4*)&kv[kvbase + 512 + (size_t)(c0 + r) * 1024 + h8];
      const __bf16* te = (const __bf16*)&t4;
#pragma unroll
      for (int j = 0; j < 8; ++j) Vt[h8 + j][r] = te[j];
    }
    cp_wait();
    __syncthreads();

    // S = Q K^T : 16 rows x 64 kv cols per wave
    v8f S[4] = {};
#pragma unroll
    for (int nt = 0; nt < 4; ++nt) {
      const __bf16* kr = &Ks[nt * 16 + lh][0];
      v16bf b0 = ld16(kr + sel * 8,      kr + 16 + sel * 8);
      v16bf b1 = ld16(kr + 32 + sel * 8, kr + 48 + sel * 8);
      S[nt] = wmma_bf16(qf[0], b0, S[nt]);
      S[nt] = wmma_bf16(qf[1], b1, S[nt]);
    }

    // online softmax (row stats via shfl within 16-lane halves)
#pragma unroll
    for (int r = 0; r < 8; ++r) {
      float cm = fmaxf(fmaxf(S[0][r], S[1][r]), fmaxf(S[2][r], S[3][r]));
#pragma unroll
      for (int m = 8; m >= 1; m >>= 1) cm = fmaxf(cm, __shfl_xor(cm, m, 32));
      const float nm = fmaxf(mrun[r], cm);
      const float sc = __expf(mrun[r] - nm);
      mrun[r] = nm;
      float cs = 0.f;
#pragma unroll
      for (int nt = 0; nt < 4; ++nt) {
        float p = __expf(S[nt][r] - nm);
        S[nt][r] = p;
        cs += p;
      }
#pragma unroll
      for (int m = 8; m >= 1; m >>= 1) cs += __shfl_xor(cs, m, 32);
      lrun[r] = lrun[r] * sc + cs;
#pragma unroll
      for (int ot = 0; ot < 4; ++ot) O[ot][r] *= sc;
      // write P row into per-wave LDS tile (C layout -> A layout via LDS)
#pragma unroll
      for (int nt = 0; nt < 4; ++nt)
        Ps[wv][sel * 8 + r][nt * 16 + lh] = (__bf16)S[nt][r];
    }
    __syncthreads();

    // O += P V : (16x64) * (64x64)
#pragma unroll
    for (int ks = 0; ks < 2; ++ks) {
      const __bf16* pr = &Ps[wv][lh][ks * 32];
      v16bf a = ld16(pr + sel * 8, pr + 16 + sel * 8);
#pragma unroll
      for (int ot = 0; ot < 4; ++ot) {
        const __bf16* vr = &Vt[ot * 16 + lh][0];
        v16bf b = ld16(vr + ks * 32 + sel * 8, vr + ks * 32 + 16 + sel * 8);
        O[ot] = wmma_bf16(a, b, O[ot]);
      }
    }
  }

  // normalize and emit bf16 (window-order rows, col = head*64 + hd)
#pragma unroll
  for (int r = 0; r < 8; ++r) {
    const float inv = 1.0f / lrun[r];
    const int g = wid * 512 + qblk * 128 + wv * 16 + sel * 8 + r;
#pragma unroll
    for (int ot = 0; ot < 4; ++ot)
      attn_out[(size_t)g * CCH + head * HDIM + ot * 16 + lh] = (__bf16)(O[ot][r] * inv);
  }
}

// ---------------- workspace layout (bytes, all 256-aligned) ----------------
#define OFF_X1    ((size_t)0)            //  32768*512*4  = 67,108,864  (x1 fp32)
#define OFF_WKV   ((size_t)67108864)     //  512*1024*2
#define OFF_WPROJ ((size_t)68157440)     //  512*512*2
#define OFF_WFC1  ((size_t)68681728)     //  512*2048*2
#define OFF_WFC2  ((size_t)70778880)     //  2048*512*2
#define OFF_XN    ((size_t)72876032)     //  32768*512*2  (xn, later reused for h)
#define OFF_KV    ((size_t)106430464)    //  32768*1024*2
#define OFF_ATTN  ((size_t)173539328)    //  32768*512*2
#define OFF_FF    ((size_t)207093760)    //  32768*2048*2

extern "C" void kernel_launch(void* const* d_in, const int* in_sizes, int n_in,
                              void* d_out, int out_size, void* d_ws, size_t ws_size,
                              hipStream_t stream) {
  const float* x    = (const float*)d_in[0];
  const float* q_ms = (const float*)d_in[1];
  const float* n1w  = (const float*)d_in[2];
  const float* n1b  = (const float*)d_in[3];
  const float* kvw  = (const float*)d_in[4];
  const float* kvb  = (const float*)d_in[5];
  const float* pw   = (const float*)d_in[6];
  const float* pb   = (const float*)d_in[7];
  const float* n2w  = (const float*)d_in[8];
  const float* n2b  = (const float*)d_in[9];
  const float* f1w  = (const float*)d_in[10];
  const float* f1b  = (const float*)d_in[11];
  const float* f2w  = (const float*)d_in[12];
  const float* f2b  = (const float*)d_in[13];

  char* ws = (char*)d_ws;
  float*  x1   = (float*)(ws + OFF_X1);
  __bf16* wkv  = (__bf16*)(ws + OFF_WKV);
  __bf16* wprj = (__bf16*)(ws + OFF_WPROJ);
  __bf16* wf1  = (__bf16*)(ws + OFF_WFC1);
  __bf16* wf2  = (__bf16*)(ws + OFF_WFC2);
  __bf16* xn   = (__bf16*)(ws + OFF_XN);       // window-order normalized x
  __bf16* hbuf = (__bf16*)(ws + OFF_XN);       // reuse for LN2 output
  __bf16* kvbf = (__bf16*)(ws + OFF_KV);
  __bf16* atb  = (__bf16*)(ws + OFF_ATTN);
  __bf16* ffb  = (__bf16*)(ws + OFF_FF);
  float*  outF = (float*)d_out;

  // 1) weights -> bf16
  cvt_bf16_kernel<<<(512 * 1024 + 255) / 256, 256, 0, stream>>>(kvw, wkv, 512 * 1024);
  cvt_bf16_kernel<<<(512 * 512 + 255) / 256, 256, 0, stream>>>(pw, wprj, 512 * 512);
  cvt_bf16_kernel<<<(512 * 2048 + 255) / 256, 256, 0, stream>>>(f1w, wf1, 512 * 2048);
  cvt_bf16_kernel<<<(2048 * 512 + 255) / 256, 256, 0, stream>>>(f2w, wf2, 2048 * 512);

  // 2) LN1 + window partition -> xn (bf16, window order)
  ln_kernel<true><<<NTOK / 8, 256, 0, stream>>>(x, n1w, n1b, xn);

  // 3) kv = xn @ kv_w + kv_b   (M=32768, N=1024, K=512)
  gemm_kernel<0><<<dim3(1024 / 128, NTOK / 128), 256, 0, stream>>>(
      xn, wkv, kvb, nullptr, nullptr, kvbf, NTOK, 1024, 512);

  // 4) attention (64 windows x 8 heads x 4 query blocks)
  attn_kernel<<<dim3(4, NHEAD, NWIN), 256, 0, stream>>>(q_ms, kvbf, atb);

  // 5) x1 = x + image(proj(attn))   (M=32768, N=512, K=512)
  gemm_kernel<1><<<dim3(512 / 128, NTOK / 128), 256, 0, stream>>>(
      atb, wprj, pb, x, x1, nullptr, NTOK, 512, 512);

  // 6) LN2 -> h (bf16, token order)
  ln_kernel<false><<<NTOK / 8, 256, 0, stream>>>(x1, n2w, n2b, hbuf);

  // 7) ff = gelu(h @ fc1_w + fc1_b)   (N=2048, K=512)
  gemm_kernel<2><<<dim3(2048 / 128, NTOK / 128), 256, 0, stream>>>(
      hbuf, wf1, f1b, nullptr, nullptr, ffb, NTOK, 2048, 512);

  // 8) out = x1 + ff @ fc2_w + fc2_b   (N=512, K=2048)
  gemm_kernel<3><<<dim3(512 / 128, NTOK / 128), 256, 0, stream>>>(
      ffb, wf2, f2b, x1, outF, nullptr, NTOK, 512, 2048);
}
